// Gate_38225208934983
// MI455X (gfx1250) — matrix operand order repaired and checked by reading, not compiled, not run on hardware
//
#include <hip/hip_runtime.h>
#include <hip/hip_bf16.h>

// ---- problem constants (from reference) ----
#define NTOK    16384
#define DIM     7168
#define NEXP    256
#define NGRP    8
#define NTOPG   4
#define TOPK    8
#define RSCALE  2.5f

// ---- tiling ----
#define MROWS   64          // token rows per workgroup (4 WMMA M tiles)
#define MT      4           // M tiles per wave
#define NT      2           // N (expert) tiles per wave -> A-fragment reuse x2
#define WAVES   8           // 8 waves x 32 experts = 256 experts
#define BLK     (WAVES * 32)
#define KT      128         // K chunk staged in LDS (4 WMMA k-steps of 32)
#define XSTR    136         // lds_x row stride in bf16 (128 + 8 pad)
#define KCH     (DIM / 32)  // 224 k-substeps over full D

typedef __attribute__((ext_vector_type(16))) __bf16 v16bf;
typedef __attribute__((ext_vector_type(8)))  float  v8f;

union Frag16 { v16bf v; int4 q[2]; __bf16 h[16]; };

// ---------------- kernel 1: W fp32 -> bf16, swizzled into paired B-fragment order ----
// Record layout: [epair 0..7][kchunk 0..223][tile 0..1][lane 0..31][16 bf16]
// expert = epair*32 + tile*16 + (lane&15);  K = kchunk*32 + 16*(lane>=16) .. +15
// -> in the hot loop a wave reads 2 KB fully contiguous per k-substep.
__global__ void __launch_bounds__(256)
wcvt_kernel(const float* __restrict__ W, __bf16* __restrict__ Wb) {
    const int t = blockIdx.x * blockDim.x + threadIdx.x;
    if (t >= (NEXP / 32) * KCH * 64) return;
    const int lane = t & 31;
    const int nt   = (t >> 5) & 1;
    const int c    = (t >> 6) % KCH;
    const int ep   = t / (64 * KCH);
    const int e    = ep * 32 + nt * 16 + (lane & 15);
    const int kb   = c * 32 + ((lane < 16) ? 0 : 16);
    const float* src = W + (size_t)e * DIM + kb;
    union { __bf16 h[16]; int4 q[2]; } u;
    #pragma unroll
    for (int q4 = 0; q4 < 4; ++q4) {
        const float4 f = ((const float4*)src)[q4];
        u.h[q4*4+0] = (__bf16)f.x; u.h[q4*4+1] = (__bf16)f.y;
        u.h[q4*4+2] = (__bf16)f.z; u.h[q4*4+3] = (__bf16)f.w;
    }
    int4* dst = (int4*)(Wb + (size_t)t * 16);
    dst[0] = u.q[0];
    dst[1] = u.q[1];
}

// ---------------- kernel 2: fused gate GEMM + routing ----------------
__global__ void __launch_bounds__(BLK)
gate_kernel(const float* __restrict__ x,
            const __bf16* __restrict__ Wb,
            const float* __restrict__ bias,
            float* __restrict__ outW,
            int*   __restrict__ outI)
{
    // double-buffered x staging; 16x256 logits tile reuses the same LDS afterwards
    __shared__ __align__(16) char smem_raw[2 * MROWS * XSTR * sizeof(__bf16)];  // 34816 B
    __shared__ float lds_b[NEXP];
    __bf16* lds_x   = (__bf16*)smem_raw;
    float*  lds_log = (float*)smem_raw;

    const int tid     = threadIdx.x;
    const int wave    = tid >> 5;
    const int lane    = tid & 31;
    const int rowBase = blockIdx.x * MROWS;

    if (tid < NEXP) lds_b[tid] = bias[tid];

    // A 16x32 layout: lanes 0-15 -> row=lane, K {0..15}; lanes 16-31 -> row=lane-16, K {16..31}
    const int arow  = lane & 15;
    const int akoff = (lane < 16) ? 0 : 16;
    // paired-swizzled B base: wave = expert-tile pair
    const __bf16* wbase = Wb + (size_t)wave * KCH * 1024 + lane * 16;

    // coop staging map: 256 threads x 32 floats = 64 rows x 128 cols per chunk
    const int xrow = tid >> 2;          // 0..63
    const int xcol = (tid & 3) * 32;    // 0,32,64,96
    const float* gx = x + (size_t)(rowBase + xrow) * DIM + xcol;
    const int lofs = xrow * XSTR + xcol;

    v8f acc[NT][MT];
    #pragma unroll
    for (int n = 0; n < NT; ++n)
        #pragma unroll
        for (int m = 0; m < MT; ++m) acc[n][m] = (v8f){};

    // prologue: stage chunk 0 into buffer 0
    {
        #pragma unroll
        for (int h = 0; h < 4; ++h) {
            union { __bf16 hh[8]; int4 q; } u;
            const float4 f0 = ((const float4*)gx)[h*2+0];
            const float4 f1 = ((const float4*)gx)[h*2+1];
            u.hh[0]=(__bf16)f0.x; u.hh[1]=(__bf16)f0.y; u.hh[2]=(__bf16)f0.z; u.hh[3]=(__bf16)f0.w;
            u.hh[4]=(__bf16)f1.x; u.hh[5]=(__bf16)f1.y; u.hh[6]=(__bf16)f1.z; u.hh[7]=(__bf16)f1.w;
            *(int4*)(lds_x + lofs + h*8) = u.q;
        }
    }

    // prologue: B fragments for the very first k-substep (software pipeline head)
    Frag16 bc[2];
    {
        const __bf16* pb = wbase;
        bc[0].q[0] = *(const int4*)(pb);
        bc[0].q[1] = *(const int4*)(pb + 8);
        bc[1].q[0] = *(const int4*)(pb + 512);
        bc[1].q[1] = *(const int4*)(pb + 520);
    }
    __syncthreads();

    int p = 0;
    for (int k0 = 0; k0 < DIM; k0 += KT) {
        const bool more = (k0 + KT) < DIM;

        // issue next chunk's x global loads early (hide behind WMMAs)
        float4 nf[8];
        if (more) {
            #pragma unroll
            for (int q4 = 0; q4 < 8; ++q4)
                nf[q4] = ((const float4*)(gx + k0 + KT))[q4];
        }

        const __bf16* xb = lds_x + p * (MROWS * XSTR);
        const int kgBase = k0 >> 5;
        #pragma unroll
        for (int kk = 0; kk < 4; ++kk) {
            const int ks = kk * 32;
            // prefetch B for the NEXT k-substep (flows across chunks/barriers)
            Frag16 bn[2];
            if (kgBase + kk + 1 < KCH) {
                const __bf16* pb = wbase + (size_t)(kgBase + kk + 1) * 1024;
                bn[0].q[0] = *(const int4*)(pb);
                bn[0].q[1] = *(const int4*)(pb + 8);
                bn[1].q[0] = *(const int4*)(pb + 512);
                bn[1].q[1] = *(const int4*)(pb + 520);
            }
            // load ALL A fragments into distinct regs -> ds_loads pipeline
            Frag16 a[MT];
            #pragma unroll
            for (int m = 0; m < MT; ++m) {
                const __bf16* pa = xb + (m * 16 + arow) * XSTR + ks + akoff;
                a[m].q[0] = *(const int4*)(pa);
                a[m].q[1] = *(const int4*)(pa + 8);
            }
            // 8 WMMAs per 4 A loads + 2 (pipelined) B loads
            #pragma unroll
            for (int m = 0; m < MT; ++m)
                acc[0][m] = __builtin_amdgcn_wmma_f32_16x16x32_bf16(
                                false, a[m].v, false, bc[0].v, (short)0, acc[0][m], false, false);
            #pragma unroll
            for (int m = 0; m < MT; ++m)
                acc[1][m] = __builtin_amdgcn_wmma_f32_16x16x32_bf16(
                                false, a[m].v, false, bc[1].v, (short)0, acc[1][m], false, false);
            bc[0] = bn[0];
            bc[1] = bn[1];
        }

        // convert + store next chunk into the other buffer
        if (more) {
            __bf16* lw = lds_x + (1 - p) * (MROWS * XSTR) + lofs;
            #pragma unroll
            for (int h = 0; h < 4; ++h) {
                union { __bf16 hh[8]; int4 q; } u;
                u.hh[0]=(__bf16)nf[h*2].x;   u.hh[1]=(__bf16)nf[h*2].y;
                u.hh[2]=(__bf16)nf[h*2].z;   u.hh[3]=(__bf16)nf[h*2].w;
                u.hh[4]=(__bf16)nf[h*2+1].x; u.hh[5]=(__bf16)nf[h*2+1].y;
                u.hh[6]=(__bf16)nf[h*2+1].z; u.hh[7]=(__bf16)nf[h*2+1].w;
                *(int4*)(lw + h*8) = u.q;
            }
        }
        __syncthreads();   // one barrier per chunk (double-buffered)
        p ^= 1;
    }

    // --- epilogue: 4 phases of 16 rows; logits tile aliases dead lds_x ---
    const int mo = (lane < 16) ? 0 : 8;             // C/D: VGPR v -> row v + mo

    #pragma unroll
    for (int m = 0; m < MT; ++m) {
        __syncthreads();                            // previous phase's reads done
        #pragma unroll
        for (int nt = 0; nt < NT; ++nt) {
            const int ncol = (wave * NT + nt) * 16 + (lane & 15);
            #pragma unroll
            for (int vv = 0; vv < 8; ++vv)
                lds_log[(vv + mo) * NEXP + ncol] = acc[nt][m][vv];
        }
        __syncthreads();

        // 8 waves route 16 rows: wave w takes rows w and w+8 (runtime loop: no acc refs)
        for (int rr = 0; rr < 2; ++rr) {
            const int r = wave + rr * 8;
            const float* Lrow = lds_log + r * NEXP;

            float sc[8];
            #pragma unroll
            for (int j = 0; j < 8; ++j) {
                const int e = lane * 8 + j;
                const float sig = 1.0f / (1.0f + __expf(-Lrow[e]));
                sc[j] = sig + lds_b[e];
            }

            // per-lane top-2 (group = lane>>2: 4 lanes x 8 elems = 32 experts/group)
            float t1 = -3.0e38f, t2 = -3.0e38f;
            #pragma unroll
            for (int j = 0; j < 8; ++j) {
                const float v = sc[j];
                if (v > t1)      { t2 = t1; t1 = v; }
                else if (v > t2) { t2 = v; }
            }
            // merge top-2 across the quad
            #pragma unroll
            for (int xm = 1; xm <= 2; xm <<= 1) {
                const float o1 = __shfl_xor(t1, xm, 32);
                const float o2 = __shfl_xor(t2, xm, 32);
                if (o1 > t1) { t2 = (t1 > o2) ? t1 : o2; t1 = o1; }
                else         { t2 = (t2 > o1) ? t2 : o1; }
            }
            const float gsum = t1 + t2;

            // rank my group among 8 group scores; keep top-4 (tie -> lower group idx)
            float gs[8];
            #pragma unroll
            for (int g = 0; g < 8; ++g) gs[g] = __shfl(gsum, g * 4, 32);
            const int myg = lane >> 2;
            int rank = 0;
            #pragma unroll
            for (int g = 0; g < 8; ++g)
                if (gs[g] > gs[myg] || (gs[g] == gs[myg] && g < myg)) rank++;
            const bool keep = (rank < NTOPG);
            #pragma unroll
            for (int j = 0; j < 8; ++j) if (!keep) sc[j] = -3.0e38f;

            // 8 rounds of wave-wide argmax (tie -> lower expert index)
            float wsum = 0.0f, myw = 0.0f;
            int myidx = 0;
            for (int t = 0; t < TOPK; ++t) {
                float lv = -3.0e38f; int lj = 0;
                #pragma unroll
                for (int j = 0; j < 8; ++j)
                    if (sc[j] > lv) { lv = sc[j]; lj = j; }
                float val = lv;
                int eidx = lane * 8 + lj;
                #pragma unroll
                for (int off = 16; off > 0; off >>= 1) {
                    const float ov = __shfl_xor(val, off, 32);
                    const int   oi = __shfl_xor(eidx, off, 32);
                    if (ov > val || (ov == val && oi < eidx)) { val = ov; eidx = oi; }
                }
                const float st = 1.0f / (1.0f + __expf(-Lrow[eidx]));  // un-biased sigmoid
                wsum += st;
                if (lane == t) { myw = st; myidx = eidx; }
                const int kl = eidx >> 3, kj = eidx & 7;
                #pragma unroll
                for (int j = 0; j < 8; ++j)
                    if (lane == kl && j == kj) sc[j] = -3.0e38f;       // remove winner
            }

            if (lane < TOPK) {
                const int row = rowBase + m * 16 + r;
                outW[(size_t)row * TOPK + lane] = myw / wsum * RSCALE;
                outI[(size_t)row * TOPK + lane] = myidx;
            }
        }
    }
}

// ---------------- launcher ----------------
extern "C" void kernel_launch(void* const* d_in, const int* in_sizes, int n_in,
                              void* d_out, int out_size, void* d_ws, size_t ws_size,
                              hipStream_t stream) {
    const float* x    = (const float*)d_in[0];   // [NTOK, DIM] fp32
    const float* W    = (const float*)d_in[1];   // [NEXP, DIM] fp32
    const float* bias = (const float*)d_in[2];   // [NEXP]      fp32

    float* outW = (float*)d_out;                                   // [NTOK, TOPK] fp32
    int*   outI = (int*)((float*)d_out + (size_t)NTOK * TOPK);     // [NTOK, TOPK] int32

    __bf16* Wb = (__bf16*)d_ws;    // NEXP*DIM*2 = 3.67 MB of workspace, pair-swizzled

    const int nThreads = (NEXP / 32) * KCH * 64;   // 114688
    wcvt_kernel<<<(nThreads + 255) / 256, 256, 0, stream>>>(W, Wb);
    gate_kernel<<<NTOK / MROWS, BLK, 0, stream>>>(x, Wb, bias, outW, outI);
}
